// CAEView_37477884625433
// MI455X (gfx1250) — compile-verified, matrix operand order checked by reading, not run on hardware
//
#include <hip/hip_runtime.h>
#include <stdint.h>

typedef __attribute__((ext_vector_type(16))) _Float16 v16h;
typedef __attribute__((ext_vector_type(8)))  float    v8f;

#define BATCH  16
#define NPTS   128
#define KNN    20
#define NEG    0.2f
#define HCATC  3072

// ============================================================================
// kNN: per-batch pairwise -||xi-xj||^2, top-20 per row (ties -> lower index),
// also counts index occurrences (for the SE mean trick).
// One block per batch, 128 threads (thread m owns column m).
// ============================================================================
__global__ void knn_kernel(const float* __restrict__ x, long long bstride, int C,
                           int* __restrict__ idx, int* __restrict__ cnt) {
  const int b = blockIdx.x;
  const int m = threadIdx.x;
  const float* xb = x + (long long)b * bstride;

  __shared__ float sxx[NPTS];
  __shared__ float sval[NPTS];
  __shared__ int   sidx[NPTS];
  __shared__ int   scnt[NPTS];

  scnt[m] = 0;
  float xx = 0.f;
  for (int c = 0; c < C; ++c) {
    float v = xb[(long long)c * NPTS + m];
    xx += v * v;
  }
  sxx[m] = xx;
  __syncthreads();

  for (int n = 0; n < NPTS; ++n) {
    float inner = 0.f;
    for (int c = 0; c < C; ++c)
      inner += xb[(long long)c * NPTS + n] * xb[(long long)c * NPTS + m];
    float val = 2.f * inner - sxx[n] - xx;   // dist[n][m]
    for (int kk = 0; kk < KNN; ++kk) {
      sval[m] = val; sidx[m] = m;
      __syncthreads();
      for (int off = 64; off > 0; off >>= 1) {
        if (m < off) {
          float v2 = sval[m + off]; int i2 = sidx[m + off];
          if (v2 > sval[m] || (v2 == sval[m] && i2 < sidx[m])) {
            sval[m] = v2; sidx[m] = i2;
          }
        }
        __syncthreads();
      }
      int win = sidx[0];
      if (m == 0) {
        idx[((long long)b * NPTS + n) * KNN + kk] = win;
        scnt[win]++;
      }
      if (m == win) val = -3.0e38f;
      __syncthreads();
    }
  }
  cnt[b * NPTS + m] = scnt[m];
}

// ============================================================================
// WMMA tile packing. f16 A-matrix (16x32) lane layout per CDNA5 ISA:
//   lane g=l>>4, r=l&15 ; VGPR v, half p :  kk = (v>=4?16:0) + g*8 + 2*(v&3) + p
// A: element [rt*16+r][kt*32+kk]  (row-major source, row stride ldw, col offset)
// B: element [kt*32+kk][nt*16+r]  (K x N source, row stride NPTS)
// Packed: tile-major, then lane*16 contiguous halves -> one 32B vector load.
// ============================================================================
__device__ __forceinline__ int kk_of(int v, int p, int g) {
  return ((v & 4) ? 16 : 0) + g * 8 + ((v & 3) << 1) + p;
}

__global__ void pack_a_kernel(const float* __restrict__ W, int ldw, int coloff,
                              _Float16* __restrict__ dst) {
  const int lane = threadIdx.x;
  const int g = lane >> 4, r = lane & 15;
  const int kt = blockIdx.x;          // C/32
  const int rt = blockIdx.y;          // O/16
  const long long tile = (long long)rt * gridDim.x + kt;
  _Float16* d = dst + tile * 512 + lane * 16;
  const float* src = W + (long long)(rt * 16 + r) * ldw + coloff + kt * 32;
  #pragma unroll
  for (int v = 0; v < 8; ++v)
    #pragma unroll
    for (int p = 0; p < 2; ++p)
      d[v * 2 + p] = (_Float16)src[kk_of(v, p, g)];
}

__global__ void pack_b_kernel(const float* __restrict__ X, long long bstride,
                              _Float16* __restrict__ dst) {
  const int lane = threadIdx.x;
  const int g = lane >> 4, r = lane & 15;
  const int nt = blockIdx.x;          // N/16
  const int kt = blockIdx.y;          // C/32
  const int b  = blockIdx.z;
  const long long tile = ((long long)b * gridDim.y + kt) * gridDim.x + nt;
  _Float16* d = dst + tile * 512 + lane * 16;
  const float* src = X + (long long)b * bstride + nt * 16 + r;
  #pragma unroll
  for (int v = 0; v < 8; ++v)
    #pragma unroll
    for (int p = 0; p < 2; ++p)
      d[v * 2 + p] = src[(long long)(kt * 32 + kk_of(v, p, g)) * NPTS];
}

// ============================================================================
// Batched GEMM: out[b,o,n] = sum_c W[o,c] * X[b,c,n]   (f16 in, f32 accum)
// Register-blocked: each wave owns a 2(O) x 4(N) grid of 16x16 WMMA tiles,
// so 6 fragment loads (192B/lane) feed 8 v_wmma ops (0.047 B/FLOP from L2).
// grid (N/64, O/128, B), block 128 (4 waves; wave w -> o-tile pair).
// ============================================================================
__global__ void gemm_wmma_kernel(const _Float16* __restrict__ pA,
                                 const _Float16* __restrict__ pB,
                                 float* __restrict__ out, int O, int C, int N) {
  const int lane = threadIdx.x & 31;
  const int w    = threadIdx.x >> 5;
  const int nb   = blockIdx.x * 4;            // first of 4 n-tiles
  const int ow   = (blockIdx.y * 4 + w) * 2;  // first of 2 o-tiles
  const int b    = blockIdx.z;
  const int KT = C >> 5, NT = N >> 4;

  v8f acc[2][4] = {};
  const _Float16* a0 = pA + ((long long)ow * KT) * 512 + lane * 16;
  const _Float16* a1 = a0 + (long long)KT * 512;
  const _Float16* bp = pB + (((long long)b * KT) * NT + nb) * 512 + lane * 16;
  const long long bstep = (long long)NT * 512;

  for (int kt = 0; kt < KT; ++kt) {
    v16h af[2], bf[4];
    af[0] = *(const v16h*)(a0);
    af[1] = *(const v16h*)(a1);
    #pragma unroll
    for (int j = 0; j < 4; ++j) bf[j] = *(const v16h*)(bp + j * 512);
    __builtin_prefetch(a0 + 512, 0, 3);          // global_prefetch_b8
    __builtin_prefetch(a1 + 512, 0, 3);
    __builtin_prefetch(bp + bstep, 0, 3);
    #pragma unroll
    for (int oi = 0; oi < 2; ++oi)
      #pragma unroll
      for (int j = 0; j < 4; ++j)
        acc[oi][j] = __builtin_amdgcn_wmma_f32_16x16x32_f16(
            false, af[oi], false, bf[j], (short)0, acc[oi][j], false, false);
    a0 += 512; a1 += 512; bp += bstep;
  }
  // C/D layout: VGPR i -> row i + (lane>>4)*8, col lane&15
  const int g = lane >> 4, r = lane & 15;
  #pragma unroll
  for (int oi = 0; oi < 2; ++oi) {
    #pragma unroll
    for (int j = 0; j < 4; ++j) {
      float* orow = out + ((long long)b * O + (ow + oi) * 16 + g * 8) * N
                  + (nb + j) * 16 + r;
      #pragma unroll
      for (int i = 0; i < 8; ++i) orow[(long long)i * N] = acc[oi][j][i];
    }
  }
}

// ============================================================================
// SE path: mean over (n,k) of bn(h) via counted sum, then 2-layer MLP.
// mean(h) = (1/NK) sum_m cnt[m]*A[m] - (1/N) sum_n A[n] + (1/N) sum_n Bv[n]
// ============================================================================
__global__ void se_mean_kernel(const float* __restrict__ A, const float* __restrict__ Bv,
                               const int* __restrict__ cnt,
                               const float* __restrict__ bs, const float* __restrict__ bb,
                               float* __restrict__ y, int O) {
  const int t = blockIdx.x * blockDim.x + threadIdx.x;
  if (t >= BATCH * O) return;
  const int b = t / O, o = t % O;
  const float* Ar = A  + ((long long)b * O + o) * NPTS;
  const float* Br = Bv + ((long long)b * O + o) * NPTS;
  const int*   cb = cnt + b * NPTS;
  float sc = 0.f, sa = 0.f, sb = 0.f;
  for (int n = 0; n < NPTS; ++n) {
    float a = Ar[n];
    sc += (float)cb[n] * a;
    sa += a;
    sb += Br[n];
  }
  float meanh = sc * (1.f / (NPTS * KNN)) + (sb - sa) * (1.f / NPTS);
  y[t] = meanh * bs[o] + bb[o];
}

__global__ void se_hidden_kernel(const float* __restrict__ y, const float* __restrict__ w1,
                                 float* __restrict__ hid, int O, int H) {
  const int t = blockIdx.x * blockDim.x + threadIdx.x;
  if (t >= BATCH * H) return;
  const int b = t / H, h = t % H;
  float acc = 0.f;
  for (int o = 0; o < O; ++o) acc += y[b * O + o] * w1[(long long)h * O + o];
  hid[t] = fmaxf(acc, 0.f);
}

__global__ void se_sig_kernel(const float* __restrict__ hid, const float* __restrict__ w2,
                              float* __restrict__ sig, int O, int H) {
  const int t = blockIdx.x * blockDim.x + threadIdx.x;
  if (t >= BATCH * O) return;
  const int b = t / O, o = t % O;
  float acc = 0.f;
  for (int h = 0; h < H; ++h) acc += hid[b * H + h] * w2[(long long)o * H + h];
  sig[t] = 1.f / (1.f + expf(-acc));
}

// ============================================================================
// Edge output: max_k lrelu( sig * (s*h + b) ), h = A[idx[n,k]] - A[n] + Bv[n],
// optional residual add, writes into the concat buffer at channel offset.
// ============================================================================
__global__ void edge_out_kernel(const float* __restrict__ A, const float* __restrict__ Bv,
                                const int* __restrict__ idx,
                                const float* __restrict__ bs, const float* __restrict__ bb,
                                const float* __restrict__ sig, const float* __restrict__ R,
                                float* __restrict__ hcat, int coff, int O) {
  const int t = blockIdx.x * blockDim.x + threadIdx.x;
  if (t >= BATCH * O * NPTS) return;
  const int n = t % NPTS;
  const int o = (t / NPTS) % O;
  const int b = t / (NPTS * O);
  const float* Ar = A + ((long long)b * O + o) * NPTS;
  const float base = Bv[((long long)b * O + o) * NPTS + n] - Ar[n];
  const int* ib = idx + ((long long)b * NPTS + n) * KNN;
  const float s = bs[o], bi = bb[o], sg = sig[b * O + o];
  float mx = -3.0e38f;
  #pragma unroll 4
  for (int k = 0; k < KNN; ++k) {
    float h = Ar[ib[k]] + base;
    float v = (h * s + bi) * sg;
    v = v >= 0.f ? v : NEG * v;
    mx = fmaxf(mx, v);
  }
  if (R) mx += R[((long long)b * O + o) * NPTS + n];
  hcat[((long long)b * HCATC + coff + o) * NPTS + n] = mx;
}

// ---- conv5 post-ops + pooling + MLP head ----------------------------------
__global__ void bn_lrelu_kernel(float* __restrict__ h, const float* __restrict__ s,
                                const float* __restrict__ b2, int O) {
  const int t = blockIdx.x * blockDim.x + threadIdx.x;
  if (t >= BATCH * O * NPTS) return;
  const int o = (t / NPTS) % O;
  float v = h[t] * s[o] + b2[o];
  h[t] = v >= 0.f ? v : NEG * v;
}

__global__ void maxmean_kernel(const float* __restrict__ h, float* __restrict__ f, int O) {
  const int t = blockIdx.x * blockDim.x + threadIdx.x;
  if (t >= BATCH * O) return;
  const int b = t / O, o = t % O;
  const float* r = h + ((long long)b * O + o) * NPTS;
  float mx = -3.0e38f, sm = 0.f;
  for (int n = 0; n < NPTS; ++n) { float v = r[n]; mx = fmaxf(mx, v); sm += v; }
  f[(long long)b * 2 * O + o]     = mx;
  f[(long long)b * 2 * O + O + o] = sm * (1.f / NPTS);
}

__global__ void linear_kernel(const float* __restrict__ in, const float* __restrict__ W,
                              const float* __restrict__ bias,
                              const float* __restrict__ bs, const float* __restrict__ bb,
                              float* __restrict__ out, int Cin, int Cout, int act) {
  const int t = blockIdx.x * blockDim.x + threadIdx.x;
  if (t >= BATCH * Cout) return;
  const int b = t / Cout, o = t % Cout;
  const float* ir = in + (long long)b * Cin;
  const float* wr = W + (long long)o * Cin;
  float acc = 0.f;
  for (int c = 0; c < Cin; ++c) acc += ir[c] * wr[c];
  if (bias) acc += bias[o];
  if (bs)   acc = acc * bs[o] + bb[o];
  if (act)  acc = acc >= 0.f ? acc : NEG * acc;
  out[t] = acc;
}

// ============================================================================
// Host side
// ============================================================================
static inline int cdiv(int a, int b) { return (a + b - 1) / b; }

struct Ws {
  float *hcat, *A, *Bv, *R, *h5, *y, *hid, *sig, *f, *f1, *f2;
  _Float16 *packW, *packX;
  int *idx, *cnt;
};

static void run_edge_block(const float* xin, long long xin_bstride, int Cin,
                           const float* convw, int O,
                           const float* bns, const float* bnb,
                           const float* sew1, const float* sew2, int H,
                           const float* resw, int coff_out,
                           const Ws& ws, hipStream_t stream) {
  const dim3 ggrid(NPTS / 64, O / 128, BATCH);
  // kNN graph on block input
  knn_kernel<<<BATCH, NPTS, 0, stream>>>(xin, xin_bstride, Cin, ws.idx, ws.cnt);
  // pack activations once (shared by A, Bv, residual GEMMs)
  pack_b_kernel<<<dim3(NPTS / 16, Cin / 32, BATCH), 32, 0, stream>>>(xin, xin_bstride, ws.packX);
  // A = W1 @ x
  pack_a_kernel<<<dim3(Cin / 32, O / 16), 32, 0, stream>>>(convw, 2 * Cin, 0, ws.packW);
  gemm_wmma_kernel<<<ggrid, 128, 0, stream>>>(ws.packW, ws.packX, ws.A, O, Cin, NPTS);
  // Bv = W2 @ x
  pack_a_kernel<<<dim3(Cin / 32, O / 16), 32, 0, stream>>>(convw, 2 * Cin, Cin, ws.packW);
  gemm_wmma_kernel<<<ggrid, 128, 0, stream>>>(ws.packW, ws.packX, ws.Bv, O, Cin, NPTS);
  // residual R = res_w @ x
  if (resw) {
    pack_a_kernel<<<dim3(Cin / 32, O / 16), 32, 0, stream>>>(resw, Cin, 0, ws.packW);
    gemm_wmma_kernel<<<ggrid, 128, 0, stream>>>(ws.packW, ws.packX, ws.R, O, Cin, NPTS);
  }
  // SE gate
  se_mean_kernel<<<cdiv(BATCH * O, 256), 256, 0, stream>>>(ws.A, ws.Bv, ws.cnt, bns, bnb, ws.y, O);
  se_hidden_kernel<<<cdiv(BATCH * H, 128), 128, 0, stream>>>(ws.y, sew1, ws.hid, O, H);
  se_sig_kernel<<<cdiv(BATCH * O, 256), 256, 0, stream>>>(ws.hid, sew2, ws.sig, O, H);
  // gather + bn + se + lrelu + max_k (+ residual) -> concat buffer
  edge_out_kernel<<<cdiv(BATCH * O * NPTS, 256), 256, 0, stream>>>(
      ws.A, ws.Bv, ws.idx, bns, bnb, ws.sig, resw ? ws.R : nullptr, ws.hcat, coff_out, O);
}

extern "C" void kernel_launch(void* const* d_in, const int* in_sizes, int n_in,
                              void* d_out, int out_size, void* d_ws, size_t ws_size,
                              hipStream_t stream) {
  (void)in_sizes; (void)n_in; (void)out_size; (void)ws_size;
  const float* x       = (const float*)d_in[0];
  const float* conv1_w = (const float*)d_in[1];
  const float* conv2_w = (const float*)d_in[2];
  const float* conv3_w = (const float*)d_in[3];
  const float* conv4_w = (const float*)d_in[4];
  const float* conv5_w = (const float*)d_in[5];
  const float* bn1_s = (const float*)d_in[6],  *bn1_b = (const float*)d_in[7];
  const float* bn2_s = (const float*)d_in[8],  *bn2_b = (const float*)d_in[9];
  const float* bn3_s = (const float*)d_in[10], *bn3_b = (const float*)d_in[11];
  const float* bn4_s = (const float*)d_in[12], *bn4_b = (const float*)d_in[13];
  const float* bn5_s = (const float*)d_in[14], *bn5_b = (const float*)d_in[15];
  const float* se1_w1 = (const float*)d_in[16], *se1_w2 = (const float*)d_in[17];
  const float* se2_w1 = (const float*)d_in[18], *se2_w2 = (const float*)d_in[19];
  const float* se3_w1 = (const float*)d_in[20], *se3_w2 = (const float*)d_in[21];
  const float* se4_w1 = (const float*)d_in[22], *se4_w2 = (const float*)d_in[23];
  const float* res1_w = (const float*)d_in[24];
  const float* res2_w = (const float*)d_in[25];
  const float* res3_w = (const float*)d_in[26];
  const float* lin1_w = (const float*)d_in[27];
  const float* bn6_s = (const float*)d_in[28], *bn6_b = (const float*)d_in[29];
  const float* lin2_w = (const float*)d_in[30], *lin2_b = (const float*)d_in[31];
  const float* bn7_s = (const float*)d_in[32], *bn7_b = (const float*)d_in[33];
  const float* lin3_w = (const float*)d_in[34], *lin3_b = (const float*)d_in[35];

  // ---- carve workspace ----
  char* base = (char*)d_ws;
  size_t off = 0;
  auto carve = [&](size_t bytes) { char* p = base + off; off += (bytes + 255) & ~(size_t)255; return p; };
  Ws ws;
  ws.hcat  = (float*)carve((size_t)BATCH * HCATC * NPTS * 4);       // 25.2 MB
  ws.A     = (float*)carve((size_t)BATCH * 1024 * NPTS * 4);        //  8.4 MB
  ws.Bv    = (float*)carve((size_t)BATCH * 1024 * NPTS * 4);
  ws.R     = (float*)carve((size_t)BATCH * 1024 * NPTS * 4);
  ws.h5    = (float*)carve((size_t)BATCH * 2048 * NPTS * 4);        // 16.8 MB
  ws.packW = (_Float16*)carve((size_t)2048 * 3072 * 2);             // 12.6 MB
  ws.packX = (_Float16*)carve((size_t)BATCH * HCATC * NPTS * 2);    // 12.6 MB
  ws.idx   = (int*)carve((size_t)BATCH * NPTS * KNN * 4);
  ws.cnt   = (int*)carve((size_t)BATCH * NPTS * 4);
  ws.y     = (float*)carve((size_t)BATCH * 2048 * 4);
  ws.hid   = (float*)carve((size_t)BATCH * 64 * 4);
  ws.sig   = (float*)carve((size_t)BATCH * 2048 * 4);
  ws.f     = (float*)carve((size_t)BATCH * 4096 * 4);
  ws.f1    = (float*)carve((size_t)BATCH * 512 * 4);
  ws.f2    = (float*)carve((size_t)BATCH * 256 * 4);

  const long long xbs = (long long)512 * NPTS;      // input batch stride
  const long long hbs = (long long)HCATC * NPTS;    // concat-buffer batch stride

  // edge blocks (x1@ch0, x2@ch512, x3@ch1024, x4@ch2048 in hcat)
  run_edge_block(x, xbs, 512, conv1_w, 512, bn1_s, bn1_b, se1_w1, se1_w2, 32,
                 nullptr, 0, ws, stream);
  run_edge_block(ws.hcat + (long long)0 * NPTS,    hbs, 512,  conv2_w, 512,
                 bn2_s, bn2_b, se2_w1, se2_w2, 32, res1_w, 512, ws, stream);
  run_edge_block(ws.hcat + (long long)512 * NPTS,  hbs, 512,  conv3_w, 1024,
                 bn3_s, bn3_b, se3_w1, se3_w2, 64, res2_w, 1024, ws, stream);
  run_edge_block(ws.hcat + (long long)1024 * NPTS, hbs, 1024, conv4_w, 1024,
                 bn4_s, bn4_b, se4_w1, se4_w2, 64, res3_w, 2048, ws, stream);

  // conv5: h5 = lrelu(bn5(conv5_w @ concat))
  pack_b_kernel<<<dim3(NPTS / 16, HCATC / 32, BATCH), 32, 0, stream>>>(ws.hcat, hbs, ws.packX);
  pack_a_kernel<<<dim3(HCATC / 32, 2048 / 16), 32, 0, stream>>>(conv5_w, HCATC, 0, ws.packW);
  gemm_wmma_kernel<<<dim3(NPTS / 64, 2048 / 128, BATCH), 128, 0, stream>>>(ws.packW, ws.packX, ws.h5, 2048, HCATC, NPTS);
  bn_lrelu_kernel<<<cdiv(BATCH * 2048 * NPTS, 256), 256, 0, stream>>>(ws.h5, bn5_s, bn5_b, 2048);

  // global max+mean pool -> f (16,4096)
  maxmean_kernel<<<cdiv(BATCH * 2048, 256), 256, 0, stream>>>(ws.h5, ws.f, 2048);

  // MLP head
  linear_kernel<<<cdiv(BATCH * 512, 256), 256, 0, stream>>>(ws.f,  lin1_w, nullptr, bn6_s, bn6_b, ws.f1, 4096, 512, 1);
  linear_kernel<<<cdiv(BATCH * 256, 256), 256, 0, stream>>>(ws.f1, lin2_w, lin2_b, bn7_s, bn7_b, ws.f2, 512, 256, 1);
  linear_kernel<<<cdiv(BATCH * 40, 256),  256, 0, stream>>>(ws.f2, lin3_w, lin3_b, nullptr, nullptr, (float*)d_out, 256, 40, 0);
}
